// VariationalLSTM_1434519076908
// MI455X (gfx1250) — matrix operand order repaired and checked by reading, not compiled, not run on hardware
//
#include <hip/hip_runtime.h>
#include <hip/hip_bf16.h>

// Bidirectional 3-layer LSTM for MI455X (gfx1250), wave32 + WMMA bf16.
//
// Phase 1 (per layer): xg = x @ W_ih^T + b  as a large bf16 WMMA GEMM
//                      (M=32768, N=2048, K=1024), f32 accumulate.
//                      Per-wave tile 32x64 -> 8 wmma per K-step; all 6
//                      fragments for a K-step are loaded up front (distinct
//                      registers) so loads clause together and WMMAs drain
//                      with staggered s_wait_loadcnt.
// Phase 2 (per layer): 512 sequential step kernels; each step does the
//                      recurrent GEMM h @ W_hh^T with v_wmma_f32_16x16x32_bf16
//                      (4 waves = 4 gates per 16x16 tile) fused with the
//                      sigmoid/tanh pointwise + length masking via LDS.
//                      The masked bf16 layer output is written straight into
//                      the next layer's GEMM A-operand buffer.
//
// Workspace requirement: ~613 MB (see layout below).

#define Bn   64
#define Tn   512
#define Hn   512
#define G4   2048          // 4*H
#define DIN  1024          // input dim == 2H for layers 1,2
#define Mrows (Bn * Tn)    // 32768

typedef __bf16 bf16_t;
typedef __attribute__((ext_vector_type(16))) __bf16 v16bf;
typedef __attribute__((ext_vector_type(8)))  __bf16 v8bf;
typedef __attribute__((ext_vector_type(8)))  float  v8f;

// ---------------- WMMA fragment loaders (contiguous-K order, wave32) --------

// A matrix 16x32 bf16: lane L -> row mbase+(L&15); K halves by (L>>4):
//   lanes 0-15 hold K = k0..k0+15, lanes 16-31 hold K = k0+16..k0+31.
__device__ inline v16bf load_a_frag(const bf16_t* __restrict__ base, int ld,
                                    int mbase, int k0, int lane) {
    const int row  = mbase + (lane & 15);
    const int koff = k0 + ((lane >> 4) << 4);         // +0 or +16
    const v8bf lo = *(const v8bf*)(base + (size_t)row * ld + koff);
    const v8bf hi = *(const v8bf*)(base + (size_t)row * ld + koff + 8);
    v16bf a;
#pragma unroll
    for (int i = 0; i < 8; ++i) { a[i] = lo[i]; a[i + 8] = hi[i]; }
    return a;
}

// B matrix 32x16 bf16: lane L -> column nbase+(L&15); 16 contiguous K values,
// lanes 0-15 hold K=k0..k0+15, lanes 16-31 hold K=k0+16..k0+31.
// Because gates = h * W^T, a B-column is a contiguous row slice of W.
__device__ inline v16bf load_b_frag(const bf16_t* __restrict__ wbase, int ld,
                                    int nbase, int k0, int lane) {
    const int wrow = nbase + (lane & 15);
    const int koff = k0 + ((lane >> 4) << 4);         // +0 or +16
    return *(const v16bf*)(wbase + (size_t)wrow * ld + koff);
}

__device__ inline v8f wmma_bf16(v16bf a, v16bf b, v8f c) {
    return __builtin_amdgcn_wmma_f32_16x16x32_bf16(
        /*neg_a=*/false, a, /*neg_b=*/false, b,
        /*c_mod=*/(short)0, c, /*reuse_a=*/false, /*reuse_b=*/false);
}

__device__ inline float sigf(float x) { return 1.0f / (1.0f + __expf(-x)); }

// ---------------- utility kernels -------------------------------------------

__global__ void k_f32_to_bf16(const float* __restrict__ src,
                              bf16_t* __restrict__ dst, size_t n) {
    size_t i = (size_t)blockIdx.x * blockDim.x + threadIdx.x;
    const size_t stride = (size_t)gridDim.x * blockDim.x;
    for (; i < n; i += stride) dst[i] = (bf16_t)src[i];
}

__global__ void k_zero_state(float* __restrict__ hf, bf16_t* __restrict__ hb,
                             float* __restrict__ cst, size_t n) {
    size_t i = (size_t)blockIdx.x * blockDim.x + threadIdx.x;
    const size_t stride = (size_t)gridDim.x * blockDim.x;
    for (; i < n; i += stride) { hf[i] = 0.0f; hb[i] = (bf16_t)0.0f; cst[i] = 0.0f; }
}

// ---------------- phase 1: xg = x @ W_ih^T + b ------------------------------
// Block: 256 threads = 8 waves in a 2(M) x 4(N) grid; each wave owns a 32x64
// C region (2x4 accumulator tiles, 8 wmma per K-step). WG tile = 64 x 256.
// Grid: (Mrows/64, 2048/256) = (512, 8).
__global__ void k_gate_gemm(const bf16_t* __restrict__ xb,   // [Mrows][DIN]
                            const bf16_t* __restrict__ wih,  // [G4][DIN]
                            const float*  __restrict__ bias, // [G4]
                            float*        __restrict__ xg) { // [Mrows][G4]
    const int lane  = threadIdx.x & 31;
    const int wave  = threadIdx.x >> 5;
    const int mb    = blockIdx.x * 64  + (wave & 1) * 32;
    const int nb    = blockIdx.y * 256 + (wave >> 1) * 64;
    const int hi16  = lane >> 4;

    v8f acc[2][4];
#pragma unroll
    for (int mi = 0; mi < 2; ++mi)
#pragma unroll
        for (int ni = 0; ni < 4; ++ni) acc[mi][ni] = (v8f){};

    for (int k0 = 0; k0 < DIN; k0 += 32) {
        if (k0 + 32 < DIN)   // speculative prefetch of next A slab (global_prefetch_b8)
            __builtin_prefetch(xb + (size_t)(mb + (lane & 15)) * DIN + k0 + 32, 0, 1);

        // Load ALL fragments for this K-step first, into distinct values,
        // so the loads form one clause and WMMAs drain with staggered waits
        // (avoids WAR serialization on shared operand registers).
        const v16bf fa0 = load_a_frag(xb, DIN, mb,      k0, lane);
        const v16bf fa1 = load_a_frag(xb, DIN, mb + 16, k0, lane);
        v16bf fb[4];
#pragma unroll
        for (int ni = 0; ni < 4; ++ni)
            fb[ni] = load_b_frag(wih, DIN, nb + ni * 16, k0, lane);

#pragma unroll
        for (int ni = 0; ni < 4; ++ni) {
            acc[0][ni] = wmma_bf16(fa0, fb[ni], acc[0][ni]);
            acc[1][ni] = wmma_bf16(fa1, fb[ni], acc[1][ni]);
        }
    }

    // C/D layout: VGPR j -> row j + 8*hi16, lane&15 -> column.
#pragma unroll
    for (int mi = 0; mi < 2; ++mi) {
#pragma unroll
        for (int ni = 0; ni < 4; ++ni) {
            const int n = nb + ni * 16 + (lane & 15);
            const float bv = bias[n];
#pragma unroll
            for (int j = 0; j < 8; ++j) {
                const int row = mb + mi * 16 + hi16 * 8 + j;
                xg[(size_t)row * G4 + n] = acc[mi][ni][j] + bv;
            }
        }
    }
}

// ---------------- phase 2: one recurrent step -------------------------------
// Grid: (128, 2) = ((B/16)*(H/16) tiles, direction).  Block: 128 = 4 waves.
// Wave g computes the 16x16 tile of gate g (K=512 -> 16 WMMAs), gates meet in
// LDS, pointwise update of (h, c) with length masking. Masked bf16 output goes
// to the next layer's A-operand buffer; final layer also writes f32 d_out.
__global__ void k_lstm_step(const float*  __restrict__ xg,      // [2][Mrows][G4]
                            const bf16_t* __restrict__ whhL,    // layer base: [2][G4][Hn]
                            const int*    __restrict__ seq_len, // [B]
                            float*        __restrict__ hf,      // [2][B][Hn]
                            bf16_t*       __restrict__ hb,      // [2][B][Hn]
                            float*        __restrict__ cst,     // [2][B][Hn]
                            bf16_t*       __restrict__ out_bf,  // [B][T][2H] next-layer input
                            float*        __restrict__ out_f32, // [B][T][2H] or nullptr
                            int s) {
    const int dir  = blockIdx.y;
    const int t    = dir ? (Tn - 1 - s) : s;
    const int bt   = blockIdx.x >> 5;          // 0..3   batch tile
    const int jt   = blockIdx.x & 31;          // 0..31  hidden tile
    const int gate = threadIdx.x >> 5;         // 0..3 -> i,f,g,o
    const int lane = threadIdx.x & 31;
    const int hi16 = lane >> 4;
    const int bbase = bt * 16;
    const int col   = gate * Hn + jt * 16;     // column base in 4H

    const bf16_t* whh  = whhL + (size_t)dir * G4 * Hn;
    const bf16_t* hbd  = hb   + (size_t)dir * Bn * Hn;
    const float*  xgd  = xg   + (size_t)dir * Mrows * G4;

    // accumulator = precomputed input gate tile (f32, C/D layout)
    v8f acc;
    {
        const int n = col + (lane & 15);
#pragma unroll
        for (int j = 0; j < 8; ++j) {
            const int b = bbase + hi16 * 8 + j;
            acc[j] = xgd[((size_t)b * Tn + t) * G4 + n];
        }
    }

    // recurrent GEMM: gates += h_prev @ W_hh^T  (K = 512)
#pragma unroll 4
    for (int k0 = 0; k0 < Hn; k0 += 32) {
        const v16bf fa = load_a_frag(hbd, Hn, bbase, k0, lane);
        const v16bf fb = load_b_frag(whh, Hn, col,   k0, lane);
        acc = wmma_bf16(fa, fb, acc);
    }

    // exchange the 4 gate tiles through LDS
    __shared__ float gsm[4][16][16];
#pragma unroll
    for (int j = 0; j < 8; ++j)
        gsm[gate][hi16 * 8 + j][lane & 15] = acc[j];
    __syncthreads();

    // pointwise: 256 elements over 128 threads (2 each)
#pragma unroll
    for (int e = 0; e < 2; ++e) {
        const int p  = threadIdx.x + e * 128;
        const int m  = p >> 4;
        const int n  = p & 15;
        const int b  = bbase + m;
        const int jg = jt * 16 + n;
        const bool valid = (t < seq_len[b]);

        const float ig = sigf(gsm[0][m][n]);
        const float fg = sigf(gsm[1][m][n]);
        const float gg = tanhf(gsm[2][m][n]);
        const float og = sigf(gsm[3][m][n]);

        const size_t sidx = ((size_t)dir * Bn + b) * Hn + jg;
        const float c_old = cst[sidx];
        const float c_new = fg * c_old + ig * gg;
        const float h_new = og * tanhf(c_new);

        const float h_keep = valid ? h_new : hf[sidx];
        cst[sidx] = valid ? c_new : c_old;
        hf[sidx]  = h_keep;
        hb[sidx]  = (bf16_t)h_keep;

        // pad_packed semantics: padded positions are zero in the layer output
        const float oval = valid ? h_new : 0.0f;
        const size_t oidx = ((size_t)b * Tn + t) * (2 * Hn) + dir * Hn + jg;
        out_bf[oidx] = (bf16_t)oval;
        if (out_f32) out_f32[oidx] = oval;
    }
}

// ---------------- host side -------------------------------------------------

extern "C" void kernel_launch(void* const* d_in, const int* in_sizes, int n_in,
                              void* d_out, int out_size, void* d_ws, size_t ws_size,
                              hipStream_t stream) {
    (void)in_sizes; (void)n_in; (void)out_size; (void)ws_size;

    const float* hidden  = (const float*)d_in[0];
    const int*   seqlen  = (const int*)  d_in[1];
    const float* w_ih[6]; const float* w_hh[6]; const float* bias[6];
    for (int p = 0; p < 6; ++p) {               // p = layer*2 + dir, order f,b
        w_ih[p] = (const float*)d_in[2 + p * 3];
        w_hh[p] = (const float*)d_in[3 + p * 3];
        bias[p] = (const float*)d_in[4 + p * 3];
    }

    // ---- workspace layout (~613 MB) ----
    char* w = (char*)d_ws;
    bf16_t* WIH  = (bf16_t*)w;                                  // 6*2048*1024 bf16 = 24 MiB
    bf16_t* WHH  = WIH + (size_t)6 * G4 * DIN;                  // 6*2048*512  bf16 = 12 MiB
    float*  XG   = (float*)(WHH + (size_t)6 * G4 * Hn);         // 2*32768*2048 f32 = 512 MiB
    bf16_t* XB   = (bf16_t*)(XG + (size_t)2 * Mrows * G4);      // 32768*1024 bf16  =  64 MiB
    float*  HF   = (float*)(XB + (size_t)Mrows * DIN);          // 2*64*512 f32
    bf16_t* HB   = (bf16_t*)(HF + (size_t)2 * Bn * Hn);         // 2*64*512 bf16
    float*  CST  = (float*)(HB + (size_t)2 * Bn * Hn);          // 2*64*512 f32

    // convert all weights to bf16 once
    for (int p = 0; p < 6; ++p) {
        k_f32_to_bf16<<<2048, 256, 0, stream>>>(w_ih[p], WIH + (size_t)p * G4 * DIN,
                                                (size_t)G4 * DIN);
        k_f32_to_bf16<<<1024, 256, 0, stream>>>(w_hh[p], WHH + (size_t)p * G4 * Hn,
                                                (size_t)G4 * Hn);
    }

    // layer-0 input -> bf16 (later layers get bf16 directly from the steps)
    k_f32_to_bf16<<<4096, 256, 0, stream>>>(hidden, XB, (size_t)Mrows * DIN);

    for (int l = 0; l < 3; ++l) {
        // xg = x @ W_ih^T + b, both directions
        for (int dir = 0; dir < 2; ++dir) {
            const int p = l * 2 + dir;
            k_gate_gemm<<<dim3(Mrows / 64, G4 / 256), 256, 0, stream>>>(
                XB, WIH + (size_t)p * G4 * DIN, bias[p],
                XG + (size_t)dir * Mrows * G4);
        }

        // reset h, c
        k_zero_state<<<256, 256, 0, stream>>>(HF, HB, CST, (size_t)2 * Bn * Hn);

        // 512 sequential recurrent steps (fwd + bwd in parallel via grid.y).
        // Steps overwrite XB in place with this layer's masked bf16 output:
        // safe because this layer's GEMMs have already fully consumed XB.
        float* outf = (l == 2) ? (float*)d_out : nullptr;
        const bf16_t* whhL = WHH + (size_t)(l * 2) * G4 * Hn;
        for (int s = 0; s < Tn; ++s) {
            k_lstm_step<<<dim3((Bn / 16) * (Hn / 16), 2), 128, 0, stream>>>(
                XG, whhL, seqlen, HF, HB, CST, XB, outf, s);
        }
    }
}
